// GNN_67027259621958
// MI455X (gfx1250) — matrix operand order repaired and checked by reading, not compiled, not run on hardware
//
#include <hip/hip_runtime.h>
#include <stdint.h>

typedef __attribute__((ext_vector_type(16))) _Float16 v16h;
typedef __attribute__((ext_vector_type(8)))  float    v8f;

#define NN 50000
#define EE 800000
#define DD 128
#define LL 3
#define NFRAG 5000
#define NGRAPH 1000
#define AS 136   // LDS row stride (halves) for A tile: 272B rows, 16B-aligned b128 frag reads
#define ERT 4    // row subtiles per edge block  -> 64 rows,  E/64  = 12500 blocks
#define NRT 5    // row subtiles per node block  -> 80 rows,  N/80  = 625 blocks

// ---------- WMMA fragment loaders (layouts per CDNA5 ISA 7.12.2) ----------

// A: 16x32 f16 tile from LDS. lane L: m = L%16, g = L/16.
// VGPR v holds K = kc*32 + (v/4)*16 + g*8 + (v%4)*2  ->  two contiguous 16B groups.
__device__ __forceinline__ v16h load_a_frag(const _Float16* Alds, int lane, int kc) {
  int m = lane & 15, g = lane >> 4;
  const uint4* p = (const uint4*)(Alds + m * AS + kc * 32 + g * 8);
  union { v16h h; uint4 q[2]; } f;
  f.q[0] = p[0];   // halves +0..+7   (v = 0..3)
  f.q[1] = p[2];   // halves +16..+23 (v = 4..7)
  return f.h;
}

// B: 32x16 f16 tile straight from global pre-transposed weights Wt[n][k] (row 128 halves).
// lane L: n = L%16, g = L/16. VGPR v holds K = kc*32 + (v/4)*16 + g*8 + (v%4)*2 -> two 16B groups.
__device__ __forceinline__ v16h load_b_frag_g(const _Float16* __restrict__ Wt,
                                              int lane, int nt, int kc) {
  int n = lane & 15, g = lane >> 4;
  const uint4* p = (const uint4*)(Wt + (((nt * 16 + n) << 7) + kc * 32 + g * 8));
  union { v16h h; uint4 q[2]; } f;
  f.q[0] = p[0];
  f.q[1] = p[2];
  return f.h;
}

// ---------- one-shot weight prep: f32 [k][n] -> f16 transposed [n][k], 9 matrices ----------
__global__ __launch_bounds__(256)
void prep_weights(const float* __restrict__ enW, const float* __restrict__ W1,
                  const float* __restrict__ W2, _Float16* __restrict__ wt) {
  size_t i = (size_t)blockIdx.x * 256 + threadIdx.x;   // 0 .. 9*16384-1
  int mat = (int)(i >> 14);                            // l*3 + {en,W1,W2}
  int idx = (int)(i & 16383);
  int k = idx >> 7, n = idx & 127;
  int l = mat / 3, w = mat - 3 * l;
  const float* src = (w == 0 ? enW : (w == 1 ? W1 : W2)) + ((size_t)l << 14);
  wt[((size_t)mat << 14) + n * 128 + k] = (_Float16)src[idx];
}

// ---------- Edge layer: E' = E @ Wl + bl ; msg = relu(x[src]+E') ; agg[dst] += msg ----------
template <bool WRITE_E>
__global__ __launch_bounds__(256)
void edge_gemm_msg(const float* __restrict__ Ein, float* __restrict__ Eout,
                   const _Float16* __restrict__ Wt, const float* __restrict__ bias,
                   const float* __restrict__ xcur, float* __restrict__ agg,
                   const int* __restrict__ srcIdx, const int* __restrict__ dstIdx) {
  __shared__ _Float16 Alds[ERT * 16 * AS];
  __shared__ int sSrc[ERT * 16], sDst[ERT * 16];
  const int tid = threadIdx.x;
  const size_t rowBase = (size_t)blockIdx.x * (ERT * 16);

  const int wave = tid >> 5, lane = tid & 31;
  v16h bf[4];                          // B fragments: direct from global (L2-hot), reused
#pragma unroll
  for (int kc = 0; kc < 4; ++kc) bf[kc] = load_b_frag_g(Wt, lane, wave, kc);

#pragma unroll
  for (int i = 0; i < (ERT * 16 * DD) / 256; ++i) {
    int idx = i * 256 + tid;           // 64x128 A tile -> f16 LDS
    int m = idx >> 7, k = idx & 127;
    Alds[m * AS + k] = (_Float16)Ein[(rowBase + m) * DD + k];
  }
  if (tid < ERT * 16) { sSrc[tid] = srcIdx[rowBase + tid]; sDst[tid] = dstIdx[rowBase + tid]; }
  __syncthreads();

  const int n = lane & 15, g = lane >> 4;
  const int col = wave * 16 + n;
  const float bcol = bias[col];

#pragma unroll
  for (int mt = 0; mt < ERT; ++mt) {
    v8f c = {};
#pragma unroll
    for (int kc = 0; kc < 4; ++kc) {
      v16h a = load_a_frag(Alds + mt * (16 * AS), lane, kc);
      c = __builtin_amdgcn_wmma_f32_16x16x32_f16(false, a, false, bf[kc], (short)0, c, false, false);
    }
#pragma unroll
    for (int v = 0; v < 8; ++v) {
      int m = mt * 16 + v + g * 8;
      float val = c[v] + bcol;
      if (WRITE_E) Eout[(rowBase + m) * DD + col] = val;
      float msg = xcur[(size_t)sSrc[m] * DD + col] + val;
      msg = msg > 0.f ? msg : 0.f;
      unsafeAtomicAdd(&agg[(size_t)sDst[m] * DD + col], msg);
    }
  }
}

// ---------- Node MLP stage 1: h1 = relu((x+agg) @ W1 + b1) ----------
__global__ __launch_bounds__(256)
void node_gemm1(const float* __restrict__ xcur, const float* __restrict__ agg,
                const _Float16* __restrict__ Wt, const float* __restrict__ bias,
                float* __restrict__ h1) {
  __shared__ _Float16 Alds[NRT * 16 * AS];
  const int tid = threadIdx.x;
  const size_t rowBase = (size_t)blockIdx.x * (NRT * 16);

  const int wave = tid >> 5, lane = tid & 31;
  v16h bf[4];
#pragma unroll
  for (int kc = 0; kc < 4; ++kc) bf[kc] = load_b_frag_g(Wt, lane, wave, kc);

#pragma unroll
  for (int i = 0; i < (NRT * 16 * DD) / 256; ++i) {
    int idx = i * 256 + tid;
    int m = idx >> 7, k = idx & 127;
    size_t gi = (rowBase + m) * DD + k;
    Alds[m * AS + k] = (_Float16)(xcur[gi] + agg[gi]);
  }
  __syncthreads();

  const int n = lane & 15, g = lane >> 4;
  const int col = wave * 16 + n;
  const float bcol = bias[col];

#pragma unroll
  for (int mt = 0; mt < NRT; ++mt) {
    v8f c = {};
#pragma unroll
    for (int kc = 0; kc < 4; ++kc) {
      v16h a = load_a_frag(Alds + mt * (16 * AS), lane, kc);
      c = __builtin_amdgcn_wmma_f32_16x16x32_f16(false, a, false, bf[kc], (short)0, c, false, false);
    }
#pragma unroll
    for (int v = 0; v < 8; ++v) {
      int m = mt * 16 + v + g * 8;
      float val = c[v] + bcol;
      h1[(rowBase + m) * DD + col] = val > 0.f ? val : 0.f;
    }
  }
}

// ---------- Node MLP stage 2: h2 = relu(h1 @ W2 + b2), accumulate BN sums ----------
__global__ __launch_bounds__(256)
void node_gemm2(const float* __restrict__ h1, const _Float16* __restrict__ Wt,
                const float* __restrict__ bias, float* __restrict__ h2,
                float* __restrict__ colsum, float* __restrict__ colsq) {
  __shared__ _Float16 Alds[NRT * 16 * AS];
  const int tid = threadIdx.x;
  const size_t rowBase = (size_t)blockIdx.x * (NRT * 16);

  const int wave = tid >> 5, lane = tid & 31;
  v16h bf[4];
#pragma unroll
  for (int kc = 0; kc < 4; ++kc) bf[kc] = load_b_frag_g(Wt, lane, wave, kc);

#pragma unroll
  for (int i = 0; i < (NRT * 16 * DD) / 256; ++i) {
    int idx = i * 256 + tid;
    int m = idx >> 7, k = idx & 127;
    Alds[m * AS + k] = (_Float16)h1[(rowBase + m) * DD + k];
  }
  __syncthreads();

  const int n = lane & 15, g = lane >> 4;
  const int col = wave * 16 + n;
  const float bcol = bias[col];

  float s = 0.f, ss = 0.f;
#pragma unroll
  for (int mt = 0; mt < NRT; ++mt) {
    v8f c = {};
#pragma unroll
    for (int kc = 0; kc < 4; ++kc) {
      v16h a = load_a_frag(Alds + mt * (16 * AS), lane, kc);
      c = __builtin_amdgcn_wmma_f32_16x16x32_f16(false, a, false, bf[kc], (short)0, c, false, false);
    }
#pragma unroll
    for (int v = 0; v < 8; ++v) {
      int m = mt * 16 + v + g * 8;
      float val = c[v] + bcol;
      val = val > 0.f ? val : 0.f;
      h2[(rowBase + m) * DD + col] = val;
      s += val; ss += val * val;
    }
  }
  // combine lane L with lane L^16 (same column, other rows)
  s  += __shfl_xor(s, 16, 32);
  ss += __shfl_xor(ss, 16, 32);
  if (lane < 16) {
    unsafeAtomicAdd(&colsum[col], s);
    unsafeAtomicAdd(&colsq[col], ss);
  }
}

// ---------- BN finalize: fold stats + gamma/beta into scale/shift, reset stats ----------
__global__ void bn_finalize(float* __restrict__ colsum, float* __restrict__ colsq,
                            const float* __restrict__ gamma, const float* __restrict__ beta,
                            float* __restrict__ scaleShift) {
  int d = threadIdx.x;   // 128 threads
  float mean = colsum[d] * (1.0f / NN);
  float var  = colsq[d] * (1.0f / NN) - mean * mean;
  float inv  = rsqrtf(var + 1e-5f);
  float sc   = gamma[d] * inv;
  scaleShift[d]       = sc;
  scaleShift[128 + d] = beta[d] - sc * mean;
  colsum[d] = 0.f; colsq[d] = 0.f;
}

// ---------- BN apply (writes next x) + zero agg for next layer ----------
__global__ __launch_bounds__(256)
void bn_apply(const float* __restrict__ h2, const float* __restrict__ scaleShift,
              float* __restrict__ xout, float* __restrict__ agg) {
  size_t i = (size_t)blockIdx.x * 256 + threadIdx.x;
  int d = (int)(i & (DD - 1));
  xout[i] = h2[i] * scaleShift[d] + scaleShift[128 + d];
  agg[i] = 0.f;
}

// ---------- init: zero agg, d_out, counts, BN stats (every launch: determinism) ----------
__global__ __launch_bounds__(256)
void init_zero(float* __restrict__ agg, float* __restrict__ out,
               float* __restrict__ cnts, float* __restrict__ stats) {
  size_t i = (size_t)blockIdx.x * 256 + threadIdx.x;
  agg[i] = 0.f;                                     // grid sized exactly N*D/256
  if (i < (size_t)(NFRAG + NGRAPH) * DD) out[i] = 0.f;
  if (i < NFRAG + NGRAPH) cnts[i] = 0.f;
  if (i < 512) stats[i] = 0.f;
}

// ---------- segment counts ----------
__global__ void seg_counts(const int* __restrict__ fb, const int* __restrict__ gb,
                           float* __restrict__ fragCnt, float* __restrict__ graphCnt) {
  int i = blockIdx.x * 256 + threadIdx.x;
  if (i < NN) {
    unsafeAtomicAdd(&fragCnt[fb[i]], 1.f);
    unsafeAtomicAdd(&graphCnt[gb[i]], 1.f);
  }
}

// ---------- sqrt-normalized sum pooling into both output sections ----------
__global__ __launch_bounds__(128)
void pool(const float* __restrict__ x, const int* __restrict__ fb, const int* __restrict__ gb,
          const float* __restrict__ fragCnt, const float* __restrict__ graphCnt,
          float* __restrict__ out) {
  int node = blockIdx.x, d = threadIdx.x;
  int f = fb[node], g = gb[node];
  float v = x[(size_t)node * DD + d];
  unsafeAtomicAdd(&out[(size_t)f * DD + d], v * rsqrtf(fragCnt[f]));
  unsafeAtomicAdd(&out[(size_t)(NFRAG + g) * DD + d], v * rsqrtf(graphCnt[g]));
}

extern "C" void kernel_launch(void* const* d_in, const int* in_sizes, int n_in,
                              void* d_out, int out_size, void* d_ws, size_t ws_size,
                              hipStream_t stream) {
  const float* x_in  = (const float*)d_in[0];
  const float* e_in  = (const float*)d_in[1];
  const float* enW   = (const float*)d_in[2];
  const float* enb   = (const float*)d_in[3];
  const float* W1    = (const float*)d_in[4];
  const float* b1    = (const float*)d_in[5];
  const float* W2    = (const float*)d_in[6];
  const float* b2    = (const float*)d_in[7];
  const float* gamma = (const float*)d_in[8];
  const float* beta  = (const float*)d_in[9];
  const int*   eidx  = (const int*)d_in[10];
  const int*   fb    = (const int*)d_in[11];
  const int*   gb    = (const int*)d_in[12];
  float* out = (float*)d_out;

  const size_t ED = (size_t)EE * DD;   // 102.4M floats
  const size_t ND = (size_t)NN * DD;   // 6.4M floats
  float* Ebuf  = (float*)d_ws;
  float* agg   = Ebuf + ED;
  float* h1    = agg + ND;
  float* h2    = h1 + ND;
  float* xbuf  = h2 + ND;
  float* stats = xbuf + ND;            // colsum[128] | colsq[128] | scaleShift[256]
  float* colsum = stats;
  float* colsq  = stats + 128;
  float* scSh   = stats + 256;
  float* cnts   = stats + 512;         // fragCnt[5000] | graphCnt[1000]
  float* fragCnt = cnts;
  float* graphCnt = cnts + NFRAG;
  _Float16* wt16 = (_Float16*)(graphCnt + NGRAPH);  // 9 x 128x128 f16, 16B-aligned

  init_zero<<<(unsigned)(ND / 256), 256, 0, stream>>>(agg, out, cnts, stats);
  prep_weights<<<(9 * 16384) / 256, 256, 0, stream>>>(enW, W1, W2, wt16);

  const float* xcur = x_in;
  const float* ecur = e_in;
  for (int l = 0; l < LL; ++l) {
    const _Float16* wtE = wt16 + ((size_t)(l * 3 + 0) << 14);
    const _Float16* wt1 = wt16 + ((size_t)(l * 3 + 1) << 14);
    const _Float16* wt2 = wt16 + ((size_t)(l * 3 + 2) << 14);

    if (l < LL - 1)
      edge_gemm_msg<true><<<EE / (ERT * 16), 256, 0, stream>>>(
          ecur, Ebuf, wtE, enb + (size_t)l * DD, xcur, agg, eidx, eidx + EE);
    else
      edge_gemm_msg<false><<<EE / (ERT * 16), 256, 0, stream>>>(
          ecur, Ebuf, wtE, enb + (size_t)l * DD, xcur, agg, eidx, eidx + EE);
    ecur = Ebuf;  // layer 1 updates edge features in place (block-local rows); layer 2 only reads

    node_gemm1<<<NN / (NRT * 16), 256, 0, stream>>>(
        xcur, agg, wt1, b1 + (size_t)l * DD, h1);
    node_gemm2<<<NN / (NRT * 16), 256, 0, stream>>>(
        h1, wt2, b2 + (size_t)l * DD, h2, colsum, colsq);
    bn_finalize<<<1, 128, 0, stream>>>(
        colsum, colsq, gamma + (size_t)l * DD, beta + (size_t)l * DD, scSh);
    bn_apply<<<(unsigned)(ND / 256), 256, 0, stream>>>(h2, scSh, xbuf, agg);
    xcur = xbuf;
  }

  seg_counts<<<(NN + 255) / 256, 256, 0, stream>>>(fb, gb, fragCnt, graphCnt);
  pool<<<NN, 128, 0, stream>>>(xbuf, fb, gb, fragCnt, graphCnt, out);
}
